// CausalSelfAttention_52647709114459
// MI455X (gfx1250) — matrix operand order repaired and checked by reading, not compiled, not run on hardware
//
#include <hip/hip_runtime.h>

// ---------------- problem constants ----------------
#define BATCH  2
#define LSEQ   2048
#define HDIM   1024
#define NHEAD  16
#define NKVH   4
#define QPK    4
#define DHEAD  64
#define SCALE  0.125f          // 1/sqrt(64)
#define MTOK   (BATCH * LSEQ)  // 4096 tokens

typedef __attribute__((ext_vector_type(16))) __bf16 v16bf;
typedef __attribute__((ext_vector_type(8)))  float  v8f;

union FragBF {           // one WMMA bf16 operand = 8 VGPRs = 32 bytes/lane
    uint4 u[2];
    v16bf v;
};

// fp32 -> bf16 (round-to-nearest-even), pure integer ops
__device__ __forceinline__ unsigned short f2bf(float f) {
    unsigned u = __builtin_bit_cast(unsigned, f);
    u += 0x7fffu + ((u >> 16) & 1u);
    return (unsigned short)(u >> 16);
}

__device__ __forceinline__ v8f vzero8() {
    v8f z;
#pragma unroll
    for (int i = 0; i < 8; ++i) z[i] = 0.0f;
    return z;
}

// A-fragment (16xK row slice, 16-bit): lane M = lane%16,
// elements 0..15 = K(16*half .. +15)
__device__ __forceinline__ v16bf load_a_bf16(const unsigned short* row, int k0, int half) {
    FragBF f;
    f.u[0] = *(const uint4*)(row + k0 + 16 * half);
    f.u[1] = *(const uint4*)(row + k0 + 16 * half + 8);
    return f.v;
}

// B-fragment (Kx16 col slice, 16-bit): lane N = lane%16,
// elements 0..15 = K(16*half .. +15), contiguous in source row
__device__ __forceinline__ v16bf load_b_bf16(const unsigned short* colrow, int k0, int half) {
    const uint4* p = (const uint4*)(colrow + k0 + 16 * half);
    FragBF f;
    f.u[0] = p[0];
    f.u[1] = p[1];
    return f.v;
}

// ---------------- fp32 -> bf16 bulk convert ----------------
__global__ __launch_bounds__(256) void cvt_f32_bf16(const float* __restrict__ src,
                                                    unsigned short* __restrict__ dst,
                                                    int n8) {
    int i = blockIdx.x * 256 + threadIdx.x;
    if (i >= n8) return;
    const float4* s = (const float4*)src;
    float4 a = s[2 * i], b = s[2 * i + 1];
    uint4 r;
    r.x = (unsigned)f2bf(a.x) | ((unsigned)f2bf(a.y) << 16);
    r.y = (unsigned)f2bf(a.z) | ((unsigned)f2bf(a.w) << 16);
    r.z = (unsigned)f2bf(b.x) | ((unsigned)f2bf(b.y) << 16);
    r.w = (unsigned)f2bf(b.z) | ((unsigned)f2bf(b.w) << 16);
    ((uint4*)dst)[i] = r;
}

// GEMM core: one wave computes a 32x64 tile of C = A(bf16) * W(bf16)^T
// 8 WMMA per k-step against 12 b128 loads; A frags reused 4x, B frags 2x.
__device__ __forceinline__ void gemm_tile_32x64(const unsigned short* __restrict__ A,
                                                const unsigned short* __restrict__ W,
                                                int K, int m0, int n0,
                                                v8f acc[2][4], int lm, int half) {
    const unsigned short* arow0 = A + (size_t)(m0 + lm) * K;
    const unsigned short* arow1 = A + (size_t)(m0 + 16 + lm) * K;
    for (int k0 = 0; k0 < K; k0 += 32) {
        v16bf a0 = load_a_bf16(arow0, k0, half);
        v16bf a1 = load_a_bf16(arow1, k0, half);
#pragma unroll
        for (int nt = 0; nt < 4; ++nt) {
            const unsigned short* wrow = W + (size_t)(n0 + nt * 16 + lm) * K;
            v16bf b = load_b_bf16(wrow, k0, half);
            acc[0][nt] = __builtin_amdgcn_wmma_f32_16x16x32_bf16(
                false, a0, false, b, (short)0, acc[0][nt], false, false);
            acc[1][nt] = __builtin_amdgcn_wmma_f32_16x16x32_bf16(
                false, a1, false, b, (short)0, acc[1][nt], false, false);
        }
    }
}

// ---------------- Q projection: Qb(bf16) = x * Wq^T ----------------
__global__ __launch_bounds__(256) void proj_q_kernel(const unsigned short* __restrict__ xb,
                                                     const unsigned short* __restrict__ wq,
                                                     unsigned short* __restrict__ Qb) {
    int wid  = blockIdx.x * 8 + (threadIdx.x >> 5);
    int lane = threadIdx.x & 31, lm = lane & 15, half = lane >> 4;
    const int NT = HDIM / 64;   // 16 column-blocks
    int m0 = (wid / NT) * 32, n0 = (wid % NT) * 64;
    v8f acc[2][4];
#pragma unroll
    for (int i = 0; i < 2; ++i)
#pragma unroll
        for (int j = 0; j < 4; ++j) acc[i][j] = vzero8();
    gemm_tile_32x64(xb, wq, HDIM, m0, n0, acc, lm, half);
#pragma unroll
    for (int mt = 0; mt < 2; ++mt)
#pragma unroll
        for (int nt = 0; nt < 4; ++nt)
#pragma unroll
            for (int r = 0; r < 8; ++r)
                Qb[(size_t)(m0 + mt * 16 + r + 8 * half) * HDIM + n0 + nt * 16 + lm] =
                    f2bf(acc[mt][nt][r]);
}

// ---------------- KV projection: scatter K row-major, V transposed ----------------
__global__ __launch_bounds__(256) void proj_kv_kernel(const unsigned short* __restrict__ xb,
                                                      const unsigned short* __restrict__ wkv,
                                                      unsigned short* __restrict__ Kb,
                                                      unsigned short* __restrict__ Vt) {
    int wid  = blockIdx.x * 8 + (threadIdx.x >> 5);
    int lane = threadIdx.x & 31, lm = lane & 15, half = lane >> 4;
    const int NKVF = 2 * HDIM / QPK;      // 512 features
    const int NT   = NKVF / 64;           // 8 column-blocks
    int m0 = (wid / NT) * 32, n0 = (wid % NT) * 64;
    v8f acc[2][4];
#pragma unroll
    for (int i = 0; i < 2; ++i)
#pragma unroll
        for (int j = 0; j < 4; ++j) acc[i][j] = vzero8();
    gemm_tile_32x64(xb, wkv, HDIM, m0, n0, acc, lm, half);
#pragma unroll
    for (int mt = 0; mt < 2; ++mt)
#pragma unroll
        for (int nt = 0; nt < 4; ++nt)
#pragma unroll
            for (int r = 0; r < 8; ++r) {
                int token = m0 + mt * 16 + r + 8 * half;
                int b = token >> 11, l = token & (LSEQ - 1);
                int f = n0 + nt * 16 + lm;
                int g = f >> 7, rem = f & 127;       // layout (g, {k|v}, d)
                unsigned short val = f2bf(acc[mt][nt][r]);
                if (rem < DHEAD)
                    Kb[((size_t)(b * NKVH + g) * LSEQ + l) * DHEAD + rem] = val;
                else
                    Vt[((size_t)(b * NKVH + g) * DHEAD + (rem - DHEAD)) * LSEQ + l] = val;
            }
}

// ---------------- flash attention, 1 wave per (b, head, 16 queries) ----------------
__global__ __launch_bounds__(32) void attn_kernel(const unsigned short* __restrict__ Qb,
                                                  const unsigned short* __restrict__ Kb,
                                                  const unsigned short* __restrict__ Vt,
                                                  unsigned short* __restrict__ Ob) {
    __shared__ unsigned int Pl[16][16];   // P^T staging: 16 q x 32 keys (bf16 pairs)

    int idx = blockIdx.x;
    int qt  = idx & (LSEQ / 16 - 1);
    int h   = (idx >> 7) & (NHEAD - 1);
    int b   = idx >> 11;
    int q0  = qt * 16;
    int g   = h >> 2;
    int lane = threadIdx.x & 31, lm = lane & 15, half = lane >> 4;
    int qglob = q0 + lm;
    float slope = exp2f(-0.5f * (float)(h + 1));   // 2^(-8/16*(h+1))

    // Q^T B-fragments over d (reused for all key blocks)
    const unsigned short* qrow = Qb + (size_t)(b * LSEQ + q0 + lm) * HDIM + h * DHEAD;
    v16bf qf[2];
#pragma unroll
    for (int kk = 0; kk < 2; ++kk) {
        FragBF f;
        const uint4* p = (const uint4*)(qrow + kk * 32 + 16 * half);
        f.u[0] = p[0];
        f.u[1] = p[1];
        qf[kk] = f.v;
    }

    const unsigned short* Kbase = Kb + (size_t)(b * NKVH + g) * LSEQ * DHEAD;
    const unsigned short* Vbase = Vt + (size_t)(b * NKVH + g) * DHEAD * LSEQ;

    v8f oacc[4];
#pragma unroll
    for (int i = 0; i < 4; ++i) oacc[i] = vzero8();
    float m_prev = -INFINITY, l_sum = 0.0f;

    for (int j0 = 0; j0 < q0 + 16; j0 += 32) {
        if (j0 + 32 < q0 + 16)
            __builtin_prefetch(Kbase + (size_t)(j0 + 32) * DHEAD, 0, 3);

        // S^T = K * Q^T : two 16-key tiles, k-dim = d (2 x 32)
        v8f s[2];
        s[0] = vzero8();
        s[1] = vzero8();
#pragma unroll
        for (int t = 0; t < 2; ++t)
#pragma unroll
            for (int kk = 0; kk < 2; ++kk) {
                const unsigned short* krow = Kbase + (size_t)(j0 + 16 * t + lm) * DHEAD + kk * 32;
                FragBF a;
                a.u[0] = *(const uint4*)(krow + 16 * half);
                a.u[1] = *(const uint4*)(krow + 16 * half + 8);
                s[t] = __builtin_amdgcn_wmma_f32_16x16x32_bf16(
                    false, a.v, false, qf[kk], (short)0, s[t], false, false);
            }

        // per-lane scores: 16 keys of one query column; ALiBi + causal mask
        float p[16];
        float lmax = -INFINITY;
#pragma unroll
        for (int t = 0; t < 2; ++t)
#pragma unroll
            for (int r = 0; r < 8; ++r) {
                int key = j0 + 16 * t + 8 * half + r;
                float v = s[t][r] * SCALE + slope * (float)(key - qglob);
                v = (key <= qglob) ? v : -1e30f;
                p[t * 8 + r] = v;
                lmax = fmaxf(lmax, v);
            }
        // online softmax update (lane pair l <-> l^16 holds same query)
        float bmax  = fmaxf(lmax, __shfl_xor(lmax, 16, 32));
        float m_new = fmaxf(m_prev, bmax);
        float alpha = __expf(m_prev - m_new);
        float lsum_loc = 0.0f;
#pragma unroll
        for (int i = 0; i < 16; ++i) {
            float e = __expf(p[i] - m_new);
            p[i] = e;
            lsum_loc += e;
        }
        l_sum  = l_sum * alpha + lsum_loc + __shfl_xor(lsum_loc, 16, 32);
        m_prev = m_new;
#pragma unroll
        for (int dt = 0; dt < 4; ++dt)
#pragma unroll
            for (int r = 0; r < 8; ++r) oacc[dt][r] *= alpha;

        // stage P^T (bf16) through LDS to get B-fragment layout
#pragma unroll
        for (int t = 0; t < 2; ++t)
#pragma unroll
            for (int i = 0; i < 4; ++i) {
                unsigned lo = f2bf(p[t * 8 + 2 * i]);
                unsigned hi = f2bf(p[t * 8 + 2 * i + 1]);
                Pl[lm][8 * t + 4 * half + i] = lo | (hi << 16);
            }
        FragBF pb;   // col = query lm, elements = keys 16*half..+15 (DS in-order per wave)
        pb.u[0] = *(const uint4*)&Pl[lm][8 * half];
        pb.u[1] = *(const uint4*)&Pl[lm][8 * half + 4];

        // O^T += V^T * P^T : 4 d-tiles, k-dim = 32 keys
#pragma unroll
        for (int dt = 0; dt < 4; ++dt) {
            const unsigned short* vrow = Vbase + (size_t)(dt * 16 + lm) * LSEQ + j0;
            FragBF a;
            a.u[0] = *(const uint4*)(vrow + 16 * half);
            a.u[1] = *(const uint4*)(vrow + 16 * half + 8);
            oacc[dt] = __builtin_amdgcn_wmma_f32_16x16x32_bf16(
                false, a.v, false, pb.v, (short)0, oacc[dt], false, false);
        }
    }

    // normalize and write O (token-major, feature = h*64 + d) as bf16
    float inv = 1.0f / l_sum;
    unsigned short* orow = Ob + (size_t)(b * LSEQ + qglob) * HDIM + h * DHEAD;
#pragma unroll
    for (int dt = 0; dt < 4; ++dt) {
        uint4 o4;
        o4.x = (unsigned)f2bf(oacc[dt][0] * inv) | ((unsigned)f2bf(oacc[dt][1] * inv) << 16);
        o4.y = (unsigned)f2bf(oacc[dt][2] * inv) | ((unsigned)f2bf(oacc[dt][3] * inv) << 16);
        o4.z = (unsigned)f2bf(oacc[dt][4] * inv) | ((unsigned)f2bf(oacc[dt][5] * inv) << 16);
        o4.w = (unsigned)f2bf(oacc[dt][6] * inv) | ((unsigned)f2bf(oacc[dt][7] * inv) << 16);
        *(uint4*)(orow + dt * 16 + 8 * half) = o4;   // 8 consecutive d values
    }
}

// ---------------- output projection: out(fp32) = O * Wo^T ----------------
__global__ __launch_bounds__(256) void proj_o_kernel(const unsigned short* __restrict__ Ob,
                                                     const unsigned short* __restrict__ wo,
                                                     float* __restrict__ out) {
    int wid  = blockIdx.x * 8 + (threadIdx.x >> 5);
    int lane = threadIdx.x & 31, lm = lane & 15, half = lane >> 4;
    const int NT = HDIM / 64;
    int m0 = (wid / NT) * 32, n0 = (wid % NT) * 64;
    v8f acc[2][4];
#pragma unroll
    for (int i = 0; i < 2; ++i)
#pragma unroll
        for (int j = 0; j < 4; ++j) acc[i][j] = vzero8();
    gemm_tile_32x64(Ob, wo, HDIM, m0, n0, acc, lm, half);
#pragma unroll
    for (int mt = 0; mt < 2; ++mt)
#pragma unroll
        for (int nt = 0; nt < 4; ++nt)
#pragma unroll
            for (int r = 0; r < 8; ++r)
                out[(size_t)(m0 + mt * 16 + r + 8 * half) * HDIM + n0 + nt * 16 + lm] =
                    acc[mt][nt][r];
}

extern "C" void kernel_launch(void* const* d_in, const int* in_sizes, int n_in,
                              void* d_out, int out_size, void* d_ws, size_t ws_size,
                              hipStream_t stream) {
    const float* x   = (const float*)d_in[0];
    const float* Wq  = (const float*)d_in[1];
    const float* Wkv = (const float*)d_in[2];
    const float* Wo  = (const float*)d_in[3];
    float* out = (float*)d_out;

    // workspace carve-up (all bf16 as ushort); total ~33 MB
    unsigned short* xb   = (unsigned short*)d_ws;
    unsigned short* wqb  = xb   + (size_t)MTOK * HDIM;
    unsigned short* wkvb = wqb  + (size_t)HDIM * HDIM;
    unsigned short* wob  = wkvb + (size_t)(2 * HDIM / QPK) * HDIM;
    unsigned short* Qb   = wob  + (size_t)HDIM * HDIM;
    unsigned short* Kb   = Qb   + (size_t)MTOK * HDIM;
    unsigned short* Vt   = Kb   + (size_t)BATCH * NKVH * LSEQ * DHEAD;
    unsigned short* Ob   = Vt   + (size_t)BATCH * NKVH * LSEQ * DHEAD;

    // 1) convert fp32 inputs/weights to bf16 once
    int nx  = MTOK * HDIM / 8;
    int nq  = HDIM * HDIM / 8;
    int nkv = (2 * HDIM / QPK) * HDIM / 8;
    cvt_f32_bf16<<<(nx  + 255) / 256, 256, 0, stream>>>(x,   xb,   nx);
    cvt_f32_bf16<<<(nq  + 255) / 256, 256, 0, stream>>>(Wq,  wqb,  nq);
    cvt_f32_bf16<<<(nkv + 255) / 256, 256, 0, stream>>>(Wkv, wkvb, nkv);
    cvt_f32_bf16<<<(nq  + 255) / 256, 256, 0, stream>>>(Wo,  wob,  nq);

    // 2) projections (waves: 32x64 output tiles, 8 waves per block)
    proj_q_kernel <<<(MTOK / 32) * (HDIM / 64) / 8, 256, 0, stream>>>(xb, wqb, Qb);
    proj_kv_kernel<<<(MTOK / 32) * ((2 * HDIM / QPK) / 64) / 8, 256, 0, stream>>>(xb, wkvb, Kb, Vt);

    // 3) flash attention
    attn_kernel<<<BATCH * NHEAD * (LSEQ / 16), 32, 0, stream>>>(Qb, Kb, Vt, Ob);

    // 4) output projection -> fp32
    proj_o_kernel<<<(MTOK / 32) * (HDIM / 64) / 8, 256, 0, stream>>>(Ob, wob, out);
}
